// TransformerBlock_79912161509415
// MI455X (gfx1250) — matrix operand order repaired and checked by reading, not compiled, not run on hardware
//
#include <hip/hip_runtime.h>
#include <hip/hip_bf16.h>

// ---------------------------------------------------------------------------
// CDNA5 (gfx1250) transformer block + top-2 routed MoE.
// - all GEMMs via v_wmma_f32_16x16x32_bf16 (f32 accum)
// - B tiles staged in LDS, double buffered, via GLOBAL_LOAD_ASYNC_TO_LDS_B128
//   (ASYNCcnt / s_wait_asynccnt) when the builtin is available.
// ---------------------------------------------------------------------------

typedef __attribute__((ext_vector_type(8)))  __bf16 v8bf;
typedef __attribute__((ext_vector_type(16))) __bf16 v16bf;
typedef __attribute__((ext_vector_type(8)))  float  v8f;
typedef __attribute__((ext_vector_type(4)))  int    v4i;

#if defined(__HIP_DEVICE_COMPILE__) &&                                   \
    __has_builtin(__builtin_amdgcn_global_load_async_to_lds_b128) &&     \
    __has_builtin(__builtin_amdgcn_s_wait_asynccnt)
#define USE_ASYNC_LDS 1
typedef __attribute__((address_space(1))) v4i* as1_v4i_p;
typedef __attribute__((address_space(3))) v4i* as3_v4i_p;
#else
#define USE_ASYNC_LDS 0
#endif

static __device__ __forceinline__ v16bf mk16(v8bf lo, v8bf hi) {
  union { v16bf v; v8bf h[2]; } u;
  u.h[0] = lo; u.h[1] = hi;
  return u.v;
}

// A-fragment (16x32, row-major source). Layout: lanes 0-15 hold rows
// M=0..15 with K = 0..15; lanes 16-31 same rows, K = 16..31.
// p = A + row*lda + k0 (row = lane&15).
static __device__ __forceinline__ v16bf ldA(const __bf16* p, int lane) {
  const int kb = (lane >> 4) * 16;
  v8bf lo = *(const v8bf*)(p + kb);
  v8bf hi = *(const v8bf*)(p + kb + 8);
  return mk16(lo, hi);
}

// B-fragment (32x16): lane holds 16 contiguous K of its column
// (col = lane&15, K-half = lane>=16).  p already includes both offsets.
static __device__ __forceinline__ v16bf ldB(const __bf16* p) {
  v8bf lo = *(const v8bf*)(p);
  v8bf hi = *(const v8bf*)(p + 8);
  return mk16(lo, hi);
}

static __device__ __forceinline__ v8f wmma_bf16(v16bf a, v16bf b, v8f c) {
  return __builtin_amdgcn_wmma_f32_16x16x32_bf16(
      false, a, false, b, (short)0, c, false, false);
}

static __device__ __forceinline__ v8f vzero8() {
  v8f z = {0.f, 0.f, 0.f, 0.f, 0.f, 0.f, 0.f, 0.f};
  return z;
}

// 16-byte global -> LDS copy, async (ASYNCcnt) when available.
static __device__ __forceinline__ void cp16_g2s(const __bf16* g, __bf16* l) {
#if USE_ASYNC_LDS
  __builtin_amdgcn_global_load_async_to_lds_b128(
      (as1_v4i_p)(g), (as3_v4i_p)(l), 0, 0);
#else
  *(v8bf*)l = *(const v8bf*)g;
#endif
}

static __device__ __forceinline__ void cp_wait_async() {
#if USE_ASYNC_LDS
  __builtin_amdgcn_s_wait_asynccnt(0);
#endif
}

// ---------------------------------------------------------------------------
// Generic 64x64-tile GEMM: 128 threads = 4 waves, each wave owns a 16-row
// stripe and four 16x16 WMMA tiles.  The shared 64(N)x32(K) B tile is staged
// in LDS (double buffered, async copy), so B global traffic is 1x not 4x.
// C-frag element (VGPR r, lane l) is row r + 8*(l>=16), col l&15.
//
// MODE 0: out bf16 = acc + bias                       (QKV projection)
// MODE 1: out f32  = acc + bias + resid               (out-proj + residual)
// MODE 2: A rows gathered via gidx, out bf16 = gelu(acc + bias)   (MoE up)
// MODE 3: out f32 += gw[row] * (acc + bias) scattered via gidx    (MoE down)
// ---------------------------------------------------------------------------
template <int MODE>
__global__ __launch_bounds__(128) void gemm64(
    const __bf16* __restrict__ A, int lda,
    const __bf16* __restrict__ Bt, int ldb,
    const float* __restrict__ bias,
    float* __restrict__ outf, __bf16* __restrict__ outb, int ldc,
    const float* __restrict__ resid, int K,
    const int* __restrict__ gidx, const float* __restrict__ gw,
    const int* __restrict__ cnt) {
  __shared__ __bf16 Bs[2][64][32];

  const int lane  = threadIdx.x & 31;
  const int wv    = threadIdx.x >> 5;
  const int rbase = blockIdx.y * 64 + wv * 16;
  const int cbase = blockIdx.x * 64;
  const int colc  = lane & 15;
  const int hi    = lane >> 4;

  int Meff = 1 << 30;
  if constexpr (MODE == 2 || MODE == 3) {
    Meff = *cnt;
    // block-uniform exit only (kernel uses barriers)
    if ((int)blockIdx.y * 64 >= Meff) return;
  }

  const int rl = rbase + colc;
  const __bf16* pA;
  if constexpr (MODE == 2) {
    const int rc = rl < Meff ? rl : (Meff - 1);
    pA = A + (size_t)gidx[rc] * lda;
  } else if constexpr (MODE == 3) {
    const int rc = rl < Meff ? rl : (Meff - 1);
    pA = A + (size_t)rc * lda;
  } else {
    pA = A + (size_t)rl * lda;
  }

  // cooperative B-tile staging: 64 rows x 32 K bf16 = 4KB = 256 x 16B chunks
  auto stage = [&](int k0, int buf) {
#pragma unroll
    for (int j = 0; j < 2; ++j) {
      const int c    = (int)threadIdx.x + 128 * j;
      const int n    = c >> 2;
      const int koff = (c & 3) * 8;
      cp16_g2s(Bt + (size_t)(cbase + n) * ldb + k0 + koff, &Bs[buf][n][koff]);
    }
  };

  v8f acc[4];
#pragma unroll
  for (int nt = 0; nt < 4; ++nt) acc[nt] = vzero8();

  stage(0, 0);
  int buf = 0;
#pragma unroll 1
  for (int k0 = 0; k0 < K; k0 += 32) {
    cp_wait_async();     // our async copies into Bs[buf] are complete
    __syncthreads();     // everyone's are, and prior reads of Bs[buf^1] done
    if (k0 + 32 < K) stage(k0 + 32, buf ^ 1);  // overlap copy with compute

    v16bf a = ldA(pA + k0, lane);
#pragma unroll
    for (int nt = 0; nt < 4; ++nt) {
      v16bf b = ldB(&Bs[buf][nt * 16 + colc][16 * hi]);
      acc[nt] = wmma_bf16(a, b, acc[nt]);
    }
    buf ^= 1;
  }

#pragma unroll
  for (int nt = 0; nt < 4; ++nt) {
    const int col = cbase + nt * 16 + colc;
    const float bs = bias[col];
#pragma unroll
    for (int r = 0; r < 8; ++r) {
      const int row = rbase + r + 8 * hi;
      const float v = acc[nt][r] + bs;
      if constexpr (MODE == 0) {
        outb[(size_t)row * ldc + col] = (__bf16)v;
      } else if constexpr (MODE == 1) {
        const size_t o = (size_t)row * ldc + col;
        outf[o] = v + resid[o];
      } else if constexpr (MODE == 2) {
        const float g = 0.5f * v * (1.0f + erff(v * 0.70710678f));
        outb[(size_t)row * ldc + col] = (__bf16)g;
      } else {
        if (row < Meff) {
          const int t = gidx[row];
          atomicAdd(&outf[(size_t)t * ldc + col], gw[row] * v);
        }
      }
    }
  }
}

// ---------------------------------------------------------------------------
// Attention: one wave per (batch, head, 16-query tile).  S=512, Dh=64.
// Pass 1: scores (2 WMMA k-steps) -> LDS f32.  Softmax per row in-wave.
// Pass 2: attn(bf16, LDS) x V via WMMA, V gathered column-wise as B^T.
// ---------------------------------------------------------------------------
__global__ __launch_bounds__(32) void attn_kernel(
    const __bf16* __restrict__ qkv, __bf16* __restrict__ ao) {
  __shared__ float  sc[16][516];
  __shared__ __bf16 at[16][512];

  const int lane = threadIdx.x & 31;
  const int colc = lane & 15;
  const int hi   = lane >> 4;
  const int qt   = blockIdx.x;   // 0..31 query tiles
  const int hd   = blockIdx.y;   // 0..11 heads
  const int b    = blockIdx.z;   // 0..31 batches
  const size_t rs = 2304;        // qkv row stride (3*D)

  const __bf16* qrow = qkv + ((size_t)b * 512 + qt * 16 + colc) * rs + hd * 64;
  const __bf16* kmat = qkv + ((size_t)b * 512) * rs + 768 + hd * 64;
  const __bf16* vmat = qkv + ((size_t)b * 512) * rs + 1536 + hd * 64;

  // ---- pass 1: scores = q k^T / 8 ----
#pragma unroll 1
  for (int kt = 0; kt < 32; ++kt) {
    v8f c = vzero8();
    const __bf16* kcol = kmat + (size_t)(kt * 16 + colc) * rs + 16 * hi;
#pragma unroll
    for (int k0 = 0; k0 < 64; k0 += 32) {
      v16bf a = ldA(qrow + k0, lane);
      v16bf bb = ldB(kcol + k0);
      c = wmma_bf16(a, bb, c);
    }
#pragma unroll
    for (int r = 0; r < 8; ++r)
      sc[r + 8 * hi][kt * 16 + colc] = c[r] * 0.125f;
  }
  __syncthreads();

  // ---- softmax over 512 keys, one row at a time (lane-parallel) ----
#pragma unroll 1
  for (int row = 0; row < 16; ++row) {
    float m = -1e30f;
    for (int j = lane; j < 512; j += 32) m = fmaxf(m, sc[row][j]);
    for (int off = 16; off; off >>= 1) m = fmaxf(m, __shfl_xor(m, off));
    float s = 0.f;
    for (int j = lane; j < 512; j += 32) {
      const float e = expf(sc[row][j] - m);
      sc[row][j] = e;
      s += e;
    }
    for (int off = 16; off; off >>= 1) s += __shfl_xor(s, off);
    const float inv = 1.0f / s;
    for (int j = lane; j < 512; j += 32)
      at[row][j] = (__bf16)(sc[row][j] * inv);
  }
  __syncthreads();

  // ---- pass 2: ao = attn x V ----
  v8f acc[4];
#pragma unroll
  for (int nt = 0; nt < 4; ++nt) acc[nt] = vzero8();

#pragma unroll 1
  for (int k0 = 0; k0 < 512; k0 += 32) {
    v16bf a = ldA(&at[colc][k0], lane);
#pragma unroll
    for (int nt = 0; nt < 4; ++nt) {
      union { v16bf v; __bf16 e[16]; } u;
      const int d = nt * 16 + colc;
#pragma unroll
      for (int i = 0; i < 16; ++i)
        u.e[i] = vmat[(size_t)(k0 + 16 * hi + i) * rs + d];
      acc[nt] = wmma_bf16(a, u.v, acc[nt]);
    }
  }

#pragma unroll
  for (int nt = 0; nt < 4; ++nt)
#pragma unroll
    for (int r = 0; r < 8; ++r) {
      const int row = qt * 16 + r + 8 * hi;
      ao[((size_t)b * 512 + row) * 768 + hd * 64 + nt * 16 + colc] =
          (__bf16)acc[nt][r];
    }
}

// ---------------------------------------------------------------------------
// LayerNorm (D=768) -> bf16, one wave per row.
// ---------------------------------------------------------------------------
__global__ __launch_bounds__(256) void ln_kernel(
    const float* __restrict__ x, const float* __restrict__ g,
    const float* __restrict__ bta, __bf16* __restrict__ out) {
  const int lane = threadIdx.x & 31;
  const size_t row = (size_t)blockIdx.x * 8 + (threadIdx.x >> 5);
  const float* xr = x + row * 768;
  float s = 0.f, s2 = 0.f;
  for (int j = lane; j < 768; j += 32) {
    const float v = xr[j];
    s += v;
    s2 += v * v;
  }
  for (int off = 16; off; off >>= 1) {
    s += __shfl_xor(s, off);
    s2 += __shfl_xor(s2, off);
  }
  const float m = s * (1.0f / 768.0f);
  const float var = s2 * (1.0f / 768.0f) - m * m;
  const float rstd = rsqrtf(var + 1e-5f);
  __bf16* orow = out + row * 768;
  for (int j = lane; j < 768; j += 32)
    orow[j] = (__bf16)((xr[j] - m) * rstd * g[j] + bta[j]);
}

// ---------------------------------------------------------------------------
// Router: one wave per token; logits (E=8), softmax, top-2, atomic list build.
// Per-token outputs are order-independent -> deterministic d_out.
// ---------------------------------------------------------------------------
__global__ __launch_bounds__(256) void router_kernel(
    const __bf16* __restrict__ xmb, const float* __restrict__ rw,
    const float* __restrict__ rb, int* __restrict__ cnt,
    int* __restrict__ idx, float* __restrict__ wts) {
  const int lane = threadIdx.x & 31;
  const int t = blockIdx.x * 8 + (threadIdx.x >> 5);
  const __bf16* xr = xmb + (size_t)t * 768;
  float a[8] = {0.f, 0.f, 0.f, 0.f, 0.f, 0.f, 0.f, 0.f};
  for (int j = lane; j < 768; j += 32) {
    const float xv = (float)xr[j];
#pragma unroll
    for (int e = 0; e < 8; ++e) a[e] += xv * rw[e * 768 + j];
  }
#pragma unroll
  for (int e = 0; e < 8; ++e)
    for (int off = 16; off; off >>= 1) a[e] += __shfl_xor(a[e], off);

  if (lane == 0) {
    float m = -1e30f;
#pragma unroll
    for (int e = 0; e < 8; ++e) {
      a[e] += rb[e];
      m = fmaxf(m, a[e]);
    }
#pragma unroll
    for (int e = 0; e < 8; ++e) a[e] = expf(a[e] - m);
    int i0 = 0;
#pragma unroll
    for (int e = 1; e < 8; ++e) if (a[e] > a[i0]) i0 = e;
    int i1 = (i0 == 0) ? 1 : 0;
#pragma unroll
    for (int e = 0; e < 8; ++e)
      if (e != i0 && a[e] > a[i1]) i1 = e;
    const float inv = 1.0f / (a[i0] + a[i1]);  // softmax denom cancels
    const float w0 = a[i0] * inv, w1 = a[i1] * inv;
    const int p0 = atomicAdd(&cnt[i0], 1);
    idx[i0 * 16384 + p0] = t;
    wts[i0 * 16384 + p0] = w0;
    const int p1 = atomicAdd(&cnt[i1], 1);
    idx[i1 * 16384 + p1] = t;
    wts[i1 * 16384 + p1] = w1;
  }
}

// ---------------------------------------------------------------------------
// fp32 -> bf16 conversions (weights), flat and transposed-per-expert.
// ---------------------------------------------------------------------------
__global__ __launch_bounds__(256) void cvt_flat(const float* __restrict__ src,
                                                __bf16* __restrict__ dst,
                                                int n) {
  const int i = blockIdx.x * 256 + threadIdx.x;
  if (i < n) dst[i] = (__bf16)src[i];
}

__global__ __launch_bounds__(256) void cvt_T(const float* __restrict__ src,
                                             __bf16* __restrict__ dst,
                                             int R, int C) {
  const size_t base = (size_t)blockIdx.z * R * C;
  const int i = blockIdx.x * 256 + threadIdx.x;
  if (i < R * C) {
    const int r = i / C, c = i % C;
    dst[base + (size_t)c * R + r] = (__bf16)src[base + i];
  }
}

__global__ void zero8(int* __restrict__ cnt) {
  if (threadIdx.x < 8) cnt[threadIdx.x] = 0;
}

// ---------------------------------------------------------------------------
extern "C" void kernel_launch(void* const* d_in, const int* in_sizes, int n_in,
                              void* d_out, int out_size, void* d_ws,
                              size_t ws_size, hipStream_t stream) {
  (void)in_sizes; (void)n_in; (void)out_size; (void)ws_size;
  // inputs (setup_inputs order)
  const float* x        = (const float*)d_in[0];
  const float* ln1_g    = (const float*)d_in[1];
  const float* ln1_b    = (const float*)d_in[2];
  const float* in_w     = (const float*)d_in[3];   // [2304,768]
  const float* in_b     = (const float*)d_in[4];
  const float* out_w    = (const float*)d_in[5];   // [768,768]
  const float* out_b    = (const float*)d_in[6];
  const float* ln2_g    = (const float*)d_in[7];
  const float* ln2_b    = (const float*)d_in[8];
  const float* rw       = (const float*)d_in[9];   // [8,768]
  const float* rb       = (const float*)d_in[10];
  const float* w1       = (const float*)d_in[11];  // [8,768,3072]
  const float* b1       = (const float*)d_in[12];
  const float* w2       = (const float*)d_in[13];  // [8,3072,768]
  const float* b2       = (const float*)d_in[14];
  float* outp           = (float*)d_out;           // [32,512,768]

  // workspace carve-up (256B aligned)
  char* p = (char*)d_ws;
  auto alloc = [&](size_t bytes) -> void* {
    void* r = (void*)p;
    p += (bytes + 255) & ~(size_t)255;
    return r;
  };
  __bf16* Wb_in  = (__bf16*)alloc((size_t)2304 * 768 * 2);
  __bf16* Wb_out = (__bf16*)alloc((size_t)768 * 768 * 2);
  __bf16* Wb1T   = (__bf16*)alloc((size_t)8 * 3072 * 768 * 2);  // [e][N=3072][K=768]
  __bf16* Wb2T   = (__bf16*)alloc((size_t)8 * 768 * 3072 * 2);  // [e][N=768][K=3072]
  __bf16* x2b    = (__bf16*)alloc((size_t)16384 * 768 * 2);
  __bf16* qkvb   = (__bf16*)alloc((size_t)16384 * 2304 * 2);
  __bf16* aob    = (__bf16*)alloc((size_t)16384 * 768 * 2);
  __bf16* xmb    = (__bf16*)alloc((size_t)16384 * 768 * 2);
  __bf16* hb     = (__bf16*)alloc((size_t)16384 * 3072 * 2);    // reused per expert
  int*    cnt    = (int*)alloc(8 * 4);
  int*    idx    = (int*)alloc((size_t)8 * 16384 * 4);
  float*  wts    = (float*)alloc((size_t)8 * 16384 * 4);

  // weight conversion (B^T layout [N,K] for contiguous WMMA B-fragments)
  cvt_flat<<<(2304 * 768 + 255) / 256, 256, 0, stream>>>(in_w, Wb_in, 2304 * 768);
  cvt_flat<<<(768 * 768 + 255) / 256, 256, 0, stream>>>(out_w, Wb_out, 768 * 768);
  cvt_T<<<dim3((768 * 3072 + 255) / 256, 1, 8), 256, 0, stream>>>(w1, Wb1T, 768, 3072);
  cvt_T<<<dim3((3072 * 768 + 255) / 256, 1, 8), 256, 0, stream>>>(w2, Wb2T, 3072, 768);

  // LN1 -> bf16
  ln_kernel<<<2048, 256, 0, stream>>>(x, ln1_g, ln1_b, x2b);

  // QKV: [16384,768] x [768,2304] (+bias) -> bf16
  gemm64<0><<<dim3(36, 256), 128, 0, stream>>>(
      x2b, 768, Wb_in, 768, in_b, nullptr, qkvb, 2304, nullptr, 768,
      nullptr, nullptr, nullptr);

  // attention
  attn_kernel<<<dim3(32, 12, 32), 32, 0, stream>>>(qkvb, aob);

  // out-proj + residual -> d_out (f32, fully written each call)
  gemm64<1><<<dim3(12, 256), 128, 0, stream>>>(
      aob, 768, Wb_out, 768, out_b, outp, nullptr, 768, x, 768,
      nullptr, nullptr, nullptr);

  // LN2 -> bf16, router, top-2 dispatch
  ln_kernel<<<2048, 256, 0, stream>>>(outp, ln2_g, ln2_b, xmb);
  zero8<<<1, 32, 0, stream>>>(cnt);
  router_kernel<<<2048, 256, 0, stream>>>(xmb, rw, rb, cnt, idx, wts);

  // routed MoE: per-expert gathered GEMM (gelu) then scaled scatter-accumulate
  for (int e = 0; e < 8; ++e) {
    gemm64<2><<<dim3(48, 256), 128, 0, stream>>>(
        xmb, 768, Wb1T + (size_t)e * 3072 * 768, 768, b1 + e * 3072,
        nullptr, hb, 3072, nullptr, 768,
        idx + e * 16384, nullptr, cnt + e);
    gemm64<3><<<dim3(12, 256), 128, 0, stream>>>(
        hb, 3072, Wb2T + (size_t)e * 768 * 3072, 3072, b2 + e * 768,
        outp, nullptr, 768, nullptr, 3072,
        idx + e * 16384, wts + e * 16384, cnt + e);
  }
}
